// GraphFakeDetector_28544352649461
// MI455X (gfx1250) — matrix operand order repaired and checked by reading, not compiled, not run on hardware
//
#include <hip/hip_runtime.h>
#include <hip/hip_bf16.h>
#include <stdint.h>

typedef __attribute__((ext_vector_type(16))) _Float16 v16h;
typedef __attribute__((ext_vector_type(8)))  float    v8f;
typedef __attribute__((ext_vector_type(4)))  unsigned int u32x4;
typedef __attribute__((ext_vector_type(8)))  int       i32x8;
typedef __attribute__((ext_vector_type(4)))  int       i32x4;

#define FD   768              // input features
#define HD   128              // hidden
#define BKK  64               // K tile
#define NKT  (FD / BKK)       // 12 K-tiles
#define NCAP 128              // ELL capacity
#define TILE_HALF  8192       // halves per packed B tile (64x128)

#if defined(__HIP_DEVICE_COMPILE__) && __has_builtin(__builtin_amdgcn_tensor_load_to_lds) && __has_builtin(__builtin_amdgcn_s_wait_tensorcnt)
#define USE_TDM 1
#else
#define USE_TDM 0
#endif

static __device__ __forceinline__ float4 ld4(const float* p) { return *(const float4*)p; }

static __device__ __forceinline__ void st4h(_Float16* dst, float4 v) {
    union { _Float16 h[4]; uint2 u; } p;
    p.h[0] = (_Float16)v.x; p.h[1] = (_Float16)v.y;
    p.h[2] = (_Float16)v.z; p.h[3] = (_Float16)v.w;
    *(uint2*)dst = p.u;   // 8B-aligned: j is a multiple of 4, lane stride 32B
}

// ---------------- graph preprocessing ----------------

__global__ void k_zero_cnt(unsigned* __restrict__ cnt, int N) {
    int i = blockIdx.x * 256 + threadIdx.x;
    if (i < N) cnt[i] = 0u;
}

__global__ void k_build_ell(const int* __restrict__ ei, int E,
                            unsigned* __restrict__ cnt, int* __restrict__ ell) {
    int e = blockIdx.x * 256 + threadIdx.x;
    if (e >= E) return;
    int s = ei[e];
    int d = ei[E + e];
    unsigned pos = atomicAdd(&cnt[d], 1u);
    if (pos < NCAP) ell[(size_t)d * NCAP + pos] = s;
}

__global__ void k_dis(const unsigned* __restrict__ cnt, float* __restrict__ dis, int N) {
    int i = blockIdx.x * 256 + threadIdx.x;
    if (i < N) dis[i] = rsqrtf((float)(cnt[i] + 1u));   // +1 self loop
}

// ---------------- one-shot: pack W1 into the WMMA B-fragment image (f16) ----------------
// Layout: [12 ktile][2 k0][8 ntile][32 lane][16 j] halves; each tile = 16KB contiguous.
// B-frag as packed here (contiguous k, same as A): lanes<16 hold K 0..15, lanes>=16 hold K 16..31; col = lane&15.

__global__ void k_pack_w1(const float* __restrict__ W1, _Float16* __restrict__ wpack) {
    int idx = blockIdx.x * 256 + threadIdx.x;      // 0 .. 768*32-1
    if (idx >= FD * (HD / 4)) return;
    int krg = idx >> 5;                            // 0..767
    int c   = (idx & 31) << 2;                     // 0..124
    float4 v = ld4(W1 + (size_t)krg * HD + c);
    int tile  = krg >> 6;
    int kin64 = krg & 63;
    int k0sel = kin64 >> 5;
    int kin   = kin64 & 31;
    int nt    = c >> 4;
    int j     = kin & 15;
    int lhi   = (kin < 16) ? 0 : 16;
    int nin   = c & 15;
    _Float16* base = wpack + ((((size_t)tile * 2 + k0sel) * 8 + nt) * 32) * 16;
    base[(size_t)(lhi + nin + 0) * 16 + j] = (_Float16)v.x;
    base[(size_t)(lhi + nin + 1) * 16 + j] = (_Float16)v.y;
    base[(size_t)(lhi + nin + 2) * 16 + j] = (_Float16)v.z;
    base[(size_t)(lhi + nin + 3) * 16 + j] = (_Float16)v.w;
}

// ---------------- GEMM1: h = x @ W1 (f16 WMMA, f32 accum) ----------------
// 128x128 block, BK=64, 8 waves. A staged two-phase (batched global loads, software-
// pipelined one tile ahead). B tiles are TDM-copied (tensor_load_to_lds) into a
// double-buffered LDS fragment image, transfer overlapped with WMMA compute.

__global__ __launch_bounds__(256) void k_gemm1(const float* __restrict__ x,
                                               const _Float16* __restrict__ wpack,
                                               float* __restrict__ h, int N) {
    __shared__ __attribute__((aligned(32))) _Float16 Apack[2][8][32][16];     // 16KB
    __shared__ __attribute__((aligned(32))) _Float16 Bbuf[2][2][8][32][16];   // 32KB

    const int tid  = threadIdx.x;
    const int wave = tid >> 5;
    const int lane = tid & 31;
    const int rowBase = blockIdx.x * 128;

    v8f acc[8];
#pragma unroll
    for (int t = 0; t < 8; ++t)
#pragma unroll
        for (int q = 0; q < 8; ++q) acc[t][q] = 0.0f;

    float4 va[8];

    auto loadA = [&](int kb) {
#pragma unroll
        for (int it = 0; it < 8; ++it) {          // all 8 loads issued before any use
            int idx = it * 256 + tid;             // 128 rows x 16 float4
            int r   = idx >> 4;
            int kr  = (idx & 15) << 2;
            int gr  = rowBase + r;
            va[it] = make_float4(0.f, 0.f, 0.f, 0.f);
            if (gr < N) va[it] = ld4(x + (size_t)gr * FD + kb + kr);
        }
    };
    auto storeA = [&]() {
#pragma unroll
        for (int it = 0; it < 8; ++it) {
            int idx = it * 256 + tid;
            int r   = idx >> 4;
            int kr  = (idx & 15) << 2;
            // A-frag as packed here (contiguous k, same as B): lanes<16 hold K 0..15, lanes>=16 hold K 16..31
            int wv = r >> 4, rin = r & 15;
            int k0sel = kr >> 5, kin = kr & 31;
            int ln = rin + ((kin >> 4) ? 16 : 0);
            int j  = kin & 15;
            st4h(&Apack[k0sel][wv][ln][j], va[it]);
        }
    };

#if USE_TDM
    auto issueB = [&](int t, int buf) {
        // 1-D TDM copy: 4096 dwords (16KB) of the packed fragment image -> LDS
        uint64_t ga  = (uint64_t)(uintptr_t)(wpack + (size_t)t * TILE_HALF);
        unsigned lds = (unsigned)(uintptr_t)&Bbuf[buf][0][0][0][0];
        u32x4 g0;
        g0[0] = 1u;                                         // count=1, user descriptor
        g0[1] = lds;                                        // lds_addr
        g0[2] = (unsigned)ga;                               // global_addr[31:0]
        g0[3] = (unsigned)((ga >> 32) & 0x1FFFFFFu) | (2u << 30);  // addr[56:32] | type=2
        i32x8 g1;
        g1[0] = (int)(2u << 16);        // workgroup_mask=0, data_size=4B
        g1[1] = (int)(4096u << 16);     // tensor_dim0 = 4096 (low16 at bit48)
        g1[2] = (int)(1u << 16);        // tensor_dim0 hi=0 | tensor_dim1 = 1
        g1[3] = (int)(4096u << 16);     // tensor_dim1 hi=0 | tile_dim0 = 4096
        g1[4] = 1;                      // tile_dim1 = 1, tile_dim2 = 0
        g1[5] = 4096;                   // tensor_dim0_stride low32
        g1[6] = 0;                      // stride hi | tensor_dim1_stride lo
        g1[7] = 0;
        i32x4 gz; gz[0] = 0; gz[1] = 0; gz[2] = 0; gz[3] = 0;
        i32x8 gz8;
#pragma unroll
        for (int q = 0; q < 8; ++q) gz8[q] = 0;
        // clang-23 (amdgpu-toolchain) 6-arg form: (g0, g1, g2, g3, g4, cpol)
        __builtin_amdgcn_tensor_load_to_lds(g0, g1, gz, gz, gz8, 0);
    };

    loadA(0);
    if (tid == 0) issueB(0, 0);
    for (int t = 0; t < NKT; ++t) {
        storeA();                                  // consumes va (tile t)
        if (t + 1 < NKT) loadA((t + 1) * BKK);     // HBM loads fly under compute(t)
        if (tid == 0) __builtin_amdgcn_s_wait_tensorcnt(0);   // B(t) landed
        __syncthreads();                           // A(t) visible; buf[(t+1)&1] free
        if (tid == 0 && t + 1 < NKT) issueB(t + 1, (t + 1) & 1);
        const int buf = t & 1;
#pragma unroll
        for (int k0 = 0; k0 < 2; ++k0) {
            v16h a = *(const v16h*)&Apack[k0][wave][lane][0];
#pragma unroll
            for (int nt = 0; nt < 8; ++nt) {
                v16h b = *(const v16h*)&Bbuf[buf][k0][nt][lane][0];
                acc[nt] = __builtin_amdgcn_wmma_f32_16x16x32_f16(
                    false, a, false, b, (short)0, acc[nt], false, false);
            }
        }
        __syncthreads();                           // protect Apack overwrite next iter
    }
#else
    loadA(0);
    for (int t = 0; t < NKT; ++t) {
        storeA();
        // stage B tile t into Bbuf[0] (contiguous copy, two-phase)
        {
            const uint4* src = (const uint4*)(wpack + (size_t)t * TILE_HALF);
            uint4* dst = (uint4*)&Bbuf[0][0][0][0][0];
            uint4 vb[4];
#pragma unroll
            for (int it = 0; it < 4; ++it) vb[it] = src[it * 256 + tid];
#pragma unroll
            for (int it = 0; it < 4; ++it) dst[it * 256 + tid] = vb[it];
        }
        if (t + 1 < NKT) loadA((t + 1) * BKK);
        __syncthreads();
#pragma unroll
        for (int k0 = 0; k0 < 2; ++k0) {
            v16h a = *(const v16h*)&Apack[k0][wave][lane][0];
#pragma unroll
            for (int nt = 0; nt < 8; ++nt) {
                v16h b = *(const v16h*)&Bbuf[0][k0][nt][lane][0];
                acc[nt] = __builtin_amdgcn_wmma_f32_16x16x32_f16(
                    false, a, false, b, (short)0, acc[nt], false, false);
            }
        }
        __syncthreads();
    }
#endif

    // epilogue: C/D 16x16 f32 layout: VGPR v, lanes<16 -> M=v, lanes>=16 -> M=v+8; N=lane&15
    int col = lane & 15;
    int rsh = (lane < 16) ? 0 : 8;
#pragma unroll
    for (int t = 0; t < 8; ++t)
#pragma unroll
        for (int v = 0; v < 8; ++v) {
            int r = rowBase + wave * 16 + rsh + v;
            if (r < N) h[(size_t)r * HD + t * 16 + col] = acc[t][v];
        }
}

// ---------------- fused: layer1 aggregation + b1 + ReLU + GEMM2 -> h2[N,2] ----------------

__global__ __launch_bounds__(256) void k_l1_agg_gemm2(
        const float* __restrict__ h, const float* __restrict__ dis,
        const int* __restrict__ ell, const unsigned* __restrict__ cnt,
        const float* __restrict__ b1, const float* __restrict__ W2,
        float* __restrict__ h2, int N) {
    int wave = threadIdx.x >> 5;
    int lane = threadIdx.x & 31;
    int i = blockIdx.x * 8 + wave;
    if (i >= N) return;

    float di = dis[i];
    float self = di * di;
    float4 a = ld4(h + (size_t)i * HD + lane * 4);
    float4 acc;
    acc.x = a.x * self; acc.y = a.y * self; acc.z = a.z * self; acc.w = a.w * self;

    unsigned c = cnt[i]; if (c > NCAP) c = NCAP;
    const int* row = ell + (size_t)i * NCAP;
    for (unsigned k = 0; k < c; ++k) {
        int s = row[k];                       // uniform across wave (broadcast)
        float nrm = dis[s] * di;
        float4 hv = ld4(h + (size_t)s * HD + lane * 4);
        acc.x += hv.x * nrm; acc.y += hv.y * nrm;
        acc.z += hv.z * nrm; acc.w += hv.w * nrm;
    }
    float4 bb = ld4(b1 + lane * 4);
    acc.x = fmaxf(acc.x + bb.x, 0.f); acc.y = fmaxf(acc.y + bb.y, 0.f);
    acc.z = fmaxf(acc.z + bb.z, 0.f); acc.w = fmaxf(acc.w + bb.w, 0.f);
    int j0 = lane * 4;
    float p0 = acc.x * W2[(j0 + 0) * 2 + 0] + acc.y * W2[(j0 + 1) * 2 + 0] +
               acc.z * W2[(j0 + 2) * 2 + 0] + acc.w * W2[(j0 + 3) * 2 + 0];
    float p1 = acc.x * W2[(j0 + 0) * 2 + 1] + acc.y * W2[(j0 + 1) * 2 + 1] +
               acc.z * W2[(j0 + 2) * 2 + 1] + acc.w * W2[(j0 + 3) * 2 + 1];
#pragma unroll
    for (int off = 16; off >= 1; off >>= 1) {
        p0 += __shfl_xor(p0, off, 32);
        p1 += __shfl_xor(p1, off, 32);
    }
    if (lane == 0) { h2[(size_t)i * 2 + 0] = p0; h2[(size_t)i * 2 + 1] = p1; }
}

// ---------------- fused: layer2 aggregation + b2 + softmax -> out[N,2] ----------------

__global__ void k_l2_agg_softmax(const float* __restrict__ h2, const float* __restrict__ dis,
                                 const int* __restrict__ ell, const unsigned* __restrict__ cnt,
                                 const float* __restrict__ b2, float* __restrict__ out, int N) {
    int i = blockIdx.x * 256 + threadIdx.x;
    if (i >= N) return;
    float di = dis[i];
    float self = di * di;
    float a0 = h2[(size_t)i * 2 + 0] * self;
    float a1 = h2[(size_t)i * 2 + 1] * self;
    unsigned c = cnt[i]; if (c > NCAP) c = NCAP;
    const int* row = ell + (size_t)i * NCAP;
    for (unsigned k = 0; k < c; ++k) {
        int s = row[k];
        float nrm = dis[s] * di;
        a0 += h2[(size_t)s * 2 + 0] * nrm;
        a1 += h2[(size_t)s * 2 + 1] * nrm;
    }
    a0 += b2[0]; a1 += b2[1];
    float m  = fmaxf(a0, a1);
    float e0 = expf(a0 - m), e1 = expf(a1 - m);
    float inv = 1.0f / (e0 + e1);
    out[(size_t)i * 2 + 0] = e0 * inv;
    out[(size_t)i * 2 + 1] = e1 * inv;
}

// ---------------- host launch ----------------

extern "C" void kernel_launch(void* const* d_in, const int* in_sizes, int n_in,
                              void* d_out, int out_size, void* d_ws, size_t ws_size,
                              hipStream_t stream) {
    (void)n_in; (void)out_size; (void)ws_size;
    const float* x  = (const float*)d_in[0];
    const int*   ei = (const int*)d_in[1];     // edge_index [2,E]
    const float* W1 = (const float*)d_in[2];
    const float* b1 = (const float*)d_in[3];
    const float* W2 = (const float*)d_in[4];
    const float* b2 = (const float*)d_in[5];
    float* out = (float*)d_out;

    const int N = in_sizes[0] / FD;
    const int E = in_sizes[1] / 2;

    char* ws = (char*)d_ws;
    size_t off = 0;
    auto carve = [&](size_t bytes) -> void* {
        void* p = ws + off;
        off += (bytes + 255) & ~(size_t)255;
        return p;
    };
    unsigned*  cnt   = (unsigned*)carve((size_t)N * 4);
    float*     dis   = (float*)carve((size_t)N * 4);
    int*       ell   = (int*)carve((size_t)N * NCAP * 4);
    float*     h     = (float*)carve((size_t)N * HD * 4);
    float*     h2    = (float*)carve((size_t)N * 2 * 4);
    _Float16*  wpack = (_Float16*)carve((size_t)FD * HD * 2);

    k_zero_cnt<<<(N + 255) / 256, 256, 0, stream>>>(cnt, N);
    k_build_ell<<<(E + 255) / 256, 256, 0, stream>>>(ei, E, cnt, ell);
    k_dis<<<(N + 255) / 256, 256, 0, stream>>>(cnt, dis, N);
    k_pack_w1<<<(FD * (HD / 4) + 255) / 256, 256, 0, stream>>>(W1, wpack);
    k_gemm1<<<(N + 127) / 128, 256, 0, stream>>>(x, wpack, h, N);
    k_l1_agg_gemm2<<<(N + 7) / 8, 256, 0, stream>>>(h, dis, ell, cnt, b1, W2, h2, N);
    k_l2_agg_softmax<<<(N + 255) / 256, 256, 0, stream>>>(h2, dis, ell, cnt, b2, out, N);
}
